// LuangAttention_79104707658595
// MI455X (gfx1250) — matrix operand order repaired and checked by reading, not compiled, not run on hardware
//
#include <hip/hip_runtime.h>
#include <math.h>

// ---------------------------------------------------------------------------
// Luong concat attention, MI455X (gfx1250, wave32, WMMA).
// dims: B=64, S=2048, D=512.  Main GEMM done in bf16 WMMA w/ f32 accumulate.
// B-matrix (W_enc) streamed through a double-buffered 2x16KB LDS stage.
// ---------------------------------------------------------------------------

typedef __bf16 bf16;
typedef __attribute__((ext_vector_type(16))) __bf16 v16bf;
typedef __attribute__((ext_vector_type(8)))  float  v8f;

#define BB 64
#define SS 2048
#define DD 512

// ---------------------------------------------------------------------------
// Kernel 1: pack W_enc (= rows D..2D-1 of W_t, [512 x 512] f32) into bf16 in
// the exact per-lane WMMA B-fragment order:
//   frag(t,ks): lane c holds B[k][n] for n = 16*t + (c&15),
//               k = 32*ks + (c>>4)*8 + (h&7) + (h>>3)*16,  h = 0..15
// Tile t occupies a contiguous 16KB block -> coalesced LDS staging later.
// ---------------------------------------------------------------------------
__global__ void pack_w_kernel(const float* __restrict__ Wt, bf16* __restrict__ wsB)
{
    int idx = blockIdx.x * 256 + threadIdx.x;       // 0 .. 512*512-1
    int h  = idx & 15;
    int c  = (idx >> 4) & 31;
    int ks = (idx >> 9) & 15;
    int t  = idx >> 13;
    int n  = t * 16 + (c & 15);
    int k  = ks * 32 + ((c >> 4) << 3) + (h & 7) + ((h >> 3) << 4);
    wsB[idx] = (bf16)Wt[(size_t)(DD + k) * DD + n];
}

// ---------------------------------------------------------------------------
// Kernel 2: pre[b][d] = b_t[d] + sum_e dec[b][e] * W_t[e][d]   (e = 0..511)
// ---------------------------------------------------------------------------
__global__ void precomp_kernel(const float* __restrict__ dec,
                               const float* __restrict__ Wt,
                               const float* __restrict__ bt,
                               float* __restrict__ pre)
{
    int idx = blockIdx.x * 256 + threadIdx.x;       // 0 .. B*D-1
    int b = idx >> 9;
    int d = idx & (DD - 1);
    float acc = bt[d];
    for (int e = 0; e < DD; ++e)
        acc += dec[b * DD + e] * Wt[(size_t)e * DD + d];
    pre[idx] = acc;
}

// ---------------------------------------------------------------------------
// Fast tanh: 1 - 2*rcp(exp(2x)+1).  v_exp_f32 + v_rcp_f32; saturates to +-1.
// ---------------------------------------------------------------------------
static __device__ inline float tanh_fast(float x)
{
    float e = __expf(2.0f * x);
    return 1.0f - 2.0f * __builtin_amdgcn_rcpf(e + 1.0f);
}

// ---------------------------------------------------------------------------
// A-fragment load+convert: per-lane WMMA A layout (ISA 7.12.2):
// halves 0..7  = K[p .. p+7], halves 8..15 = K[p+16 .. p+23]
// ---------------------------------------------------------------------------
static __device__ inline v16bf load_cvt_afrag(const float* __restrict__ p)
{
    float4 x = ((const float4*)p)[0];
    float4 y = ((const float4*)p)[1];
    float4 z = ((const float4*)(p + 16))[0];
    float4 w = ((const float4*)(p + 16))[1];
    v16bf v;
    v[0]=(bf16)x.x;  v[1]=(bf16)x.y;  v[2]=(bf16)x.z;  v[3]=(bf16)x.w;
    v[4]=(bf16)y.x;  v[5]=(bf16)y.y;  v[6]=(bf16)y.z;  v[7]=(bf16)y.w;
    v[8]=(bf16)z.x;  v[9]=(bf16)z.y;  v[10]=(bf16)z.z; v[11]=(bf16)z.w;
    v[12]=(bf16)w.x; v[13]=(bf16)w.y; v[14]=(bf16)w.z; v[15]=(bf16)w.w;
    return v;
}

// ---------------------------------------------------------------------------
// Kernel 3: fused  score[b,s] = v_a . tanh(enc[b,s,:] @ W_enc + pre[b,:])
// 128 threads = 4 waves; wave owns 32 rows (two 16-row A tiles), A fragments
// register-resident (enc read exactly once).  B tiles (16KB each) stream
// through a double-buffered LDS stage shared by all 4 waves; the next
// tile's global loads are in flight while the current tile computes.
// ---------------------------------------------------------------------------
__global__ void __launch_bounds__(128, 1)
score_kernel(const float* __restrict__ enc,
             const bf16*  __restrict__ wsB,
             const float* __restrict__ pre,
             const float* __restrict__ va,
             float*       __restrict__ scores)
{
    __shared__ uint4 Bbuf[2][1024];              // 2 x 16KB B-tile stage

    const int tid  = threadIdx.x;
    const int b    = blockIdx.x >> 4;            // 16 row-blocks per batch
    const int rblk = (blockIdx.x & 15) * 128;    // 128 rows per workgroup
    const int lane = tid & 31;
    const int w    = tid >> 5;
    const int g    = lane >> 4;                  // K-half group
    const int l    = lane & 15;                  // row (A) / col (B/C) in tile

    const int rowBase = rblk + w * 32;           // this wave's 32 rows
    const float* encW = enc + ((size_t)b * SS + rowBase) * DD;
    const uint4* gB   = (const uint4*)wsB;       // tile t = gB[t*1024 ..]

    // ---- load & convert all A fragments into registers (enc read once) ----
    v16bf afr[2][16];
#pragma unroll
    for (int set = 0; set < 2; ++set) {
        const float* rowp = encW + (size_t)(set * 16 + l) * DD;
#pragma unroll
        for (int ks = 0; ks < 16; ++ks)
            afr[set][ks] = load_cvt_afrag(rowp + ks * 32 + 8 * g);
    }

    // ---- pipeline prologue: tile0 -> LDS buf0, tile1 -> regs ---------------
    uint4 stage[8];
#pragma unroll
    for (int i = 0; i < 8; ++i) stage[i] = gB[tid * 8 + i];
#pragma unroll
    for (int i = 0; i < 8; ++i) Bbuf[0][tid * 8 + i] = stage[i];
#pragma unroll
    for (int i = 0; i < 8; ++i) stage[i] = gB[1024 + tid * 8 + i];
    __syncthreads();

    float s0[8], s1[8];
#pragma unroll
    for (int r = 0; r < 8; ++r) { s0[r] = 0.f; s1[r] = 0.f; }

    // ---- N-tile loop: GEMM-accumulate then fused tanh/v_a epilogue --------
#pragma unroll 1
    for (int t = 0; t < 32; ++t) {
        const uint4* bb = &Bbuf[t & 1][0];

        v8f a0, a1;
#pragma unroll
        for (int i = 0; i < 8; ++i) { a0[i] = 0.f; a1[i] = 0.f; }

#pragma unroll
        for (int ks = 0; ks < 16; ++ks) {
            union { uint4 u[2]; v16bf v; } bf_;
            const int fo = 2 * (ks * 32 + lane);
            bf_.u[0] = bb[fo];
            bf_.u[1] = bb[fo + 1];
            a0 = __builtin_amdgcn_wmma_f32_16x16x32_bf16(
                     false, afr[0][ks], false, bf_.v, (short)0, a0, false, false);
            a1 = __builtin_amdgcn_wmma_f32_16x16x32_bf16(
                     false, afr[1][ks], false, bf_.v, (short)0, a1, false, false);
        }

        const int n  = t * 16 + l;               // output-feature index
        const float pv = pre[b * DD + n];
        const float vv = va[n];
#pragma unroll
        for (int r = 0; r < 8; ++r) {
            s0[r] += vv * tanh_fast(a0[r] + pv);
            s1[r] += vv * tanh_fast(a1[r] + pv);
        }

        // ---- stage tile t+1 into the other LDS buffer, prefetch t+2 -------
        if (t < 31) {
            uint4* dst = &Bbuf[(t + 1) & 1][0];
#pragma unroll
            for (int i = 0; i < 8; ++i) dst[tid * 8 + i] = stage[i];
            if (t < 30) {
#pragma unroll
                for (int i = 0; i < 8; ++i)
                    stage[i] = gB[(size_t)(t + 2) * 1024 + tid * 8 + i];
            }
            __syncthreads();
        }
    }

    // ---- reduce partial scores across the 16 n-lanes of each half-wave ----
#pragma unroll
    for (int r = 0; r < 8; ++r) {
        for (int m = 1; m < 16; m <<= 1) {
            s0[r] += __shfl_xor(s0[r], m, 32);
            s1[r] += __shfl_xor(s1[r], m, 32);
        }
    }
    if (l == 0) {   // lanes 0 and 16 write: C-tile row m = 8*g + r
#pragma unroll
        for (int r = 0; r < 8; ++r) {
            scores[b * SS + rowBase +      8 * g + r] = s0[r];
            scores[b * SS + rowBase + 16 + 8 * g + r] = s1[r];
        }
    }
}

// ---------------------------------------------------------------------------
// Kernel 4: softmax over S per batch -> alignment output (b_v dropped: softmax
// is shift-invariant).
// ---------------------------------------------------------------------------
__global__ void softmax_kernel(const float* __restrict__ scores,
                               float* __restrict__ align)
{
    const int b = blockIdx.x, tid = threadIdx.x;
    __shared__ float red[256];
    const float* sc = scores + b * SS;
    float* oa = align + b * SS;

    float mx = -INFINITY;
    for (int i = tid; i < SS; i += 256) mx = fmaxf(mx, sc[i]);
    red[tid] = mx; __syncthreads();
    for (int o = 128; o > 0; o >>= 1) {
        if (tid < o) red[tid] = fmaxf(red[tid], red[tid + o]);
        __syncthreads();
    }
    mx = red[0]; __syncthreads();

    float sum = 0.f;
    for (int i = tid; i < SS; i += 256) {
        float e = __expf(sc[i] - mx);
        oa[i] = e;
        sum += e;
    }
    red[tid] = sum; __syncthreads();
    for (int o = 128; o > 0; o >>= 1) {
        if (tid < o) red[tid] += red[tid + o];
        __syncthreads();
    }
    const float inv = 1.f / red[0];
    for (int i = tid; i < SS; i += 256) oa[i] *= inv;
}

// ---------------------------------------------------------------------------
// Kernel 5: context[b,d] = sum_s align[b,s] * enc[b,s,d]  (coalesced over d)
// ---------------------------------------------------------------------------
__global__ void context_kernel(const float* __restrict__ enc,
                               const float* __restrict__ align,
                               float* __restrict__ ctx)
{
    const int b = blockIdx.x, d = threadIdx.x;   // 512 threads
    __shared__ float al[SS];
    for (int i = threadIdx.x; i < SS; i += DD) al[i] = align[b * SS + i];
    __syncthreads();

    const float* e = enc + (size_t)b * SS * DD + d;
    float acc = 0.f;
#pragma unroll 4
    for (int s = 0; s < SS; ++s) acc += al[s] * e[(size_t)s * DD];
    ctx[b * DD + d] = acc;
}

// ---------------------------------------------------------------------------
extern "C" void kernel_launch(void* const* d_in, const int* in_sizes, int n_in,
                              void* d_out, int out_size, void* d_ws, size_t ws_size,
                              hipStream_t stream)
{
    (void)in_sizes; (void)n_in; (void)out_size; (void)ws_size;
    const float* dec = (const float*)d_in[0];   // [B,1,D]
    const float* enc = (const float*)d_in[1];   // [B,S,D]
    const float* Wt  = (const float*)d_in[2];   // [2D,D]
    const float* bt  = (const float*)d_in[3];   // [D]
    const float* va  = (const float*)d_in[4];   // [D,1]
    // d_in[5] = b_v : softmax shift-invariant, unused.

    // workspace layout: [0,512K) packed bf16 W_enc | [512K,640K) pre | [640K,..) scores
    bf16*  wsB   = (bf16*)d_ws;
    float* wsPre = (float*)((char*)d_ws + 512 * 1024);
    float* wsSc  = (float*)((char*)d_ws + 640 * 1024);

    float* outCtx   = (float*)d_out;            // [B,1,D]
    float* outAlign = outCtx + BB * DD;         // [B,1,S]

    pack_w_kernel   <<<(DD * DD) / 256, 256, 0, stream>>>(Wt, wsB);
    precomp_kernel  <<<(BB * DD) / 256, 256, 0, stream>>>(dec, Wt, bt, wsPre);
    score_kernel    <<<BB * (SS / 128), 128, 0, stream>>>(enc, wsB, wsPre, va, wsSc);
    softmax_kernel  <<<BB, 256, 0, stream>>>(wsSc, outAlign);
    context_kernel  <<<BB, DD, 0, stream>>>(enc, outAlign, outCtx);
}
